// SS2D_Mamba_35150012351233
// MI455X (gfx1250) — matrix operand order repaired and checked by reading, not compiled, not run on hardware
//
#include <hip/hip_runtime.h>
#include <hip/hip_bf16.h>
#include <cstdint>
#include <cstddef>

// ---- problem constants -----------------------------------------------------
#define CDIM     192
#define DINNER   384
#define DSTATE   16
#define DTRANK   12
#define LSEQ     4096
#define NBATCH   4
#define BLTOT    (NBATCH * LSEQ)        // 16384 positions
#define PROJN    44                     // dt_rank + 2*d_state
#define NCHUNK   16
#define CHUNK    256                    // NCHUNK*CHUNK == LSEQ

typedef __attribute__((ext_vector_type(16))) __bf16 v16bf;
typedef __attribute__((ext_vector_type(8)))  float  v8f;
typedef unsigned int u32x4 __attribute__((ext_vector_type(4)));
typedef int          i32x4 __attribute__((ext_vector_type(4)));
typedef int          i32x8 __attribute__((ext_vector_type(8)));

#if __has_builtin(__builtin_amdgcn_tensor_load_to_lds)
#define HAVE_TDM 1
#else
#define HAVE_TDM 0
#endif

// ---- TDM: async 2D tile (f32) global -> LDS --------------------------------
// tensorW: valid columns from gaddr (OOB columns read as zero)
// tileW/tileH: tile to move; rowStrideElems: elements between rows.
__device__ __forceinline__ void tdm_load_2d_f32(const void* gaddr, unsigned ldsOff,
                                                unsigned tensorW, unsigned tensorH,
                                                unsigned tileW, unsigned tileH,
                                                unsigned rowStrideElems)
{
#if HAVE_TDM
    unsigned long long ga = (unsigned long long)gaddr;
    u32x4 g0;
    g0[0] = 1u;                                                   // count=1, user mode
    g0[1] = ldsOff;                                               // lds_addr (bytes)
    g0[2] = (unsigned)ga;                                         // global_addr[31:0]
    g0[3] = (unsigned)((ga >> 32) & 0x01FFFFFFull) | (2u << 30);  // ga[56:32] | type=2
    i32x8 g1;
    g1[0] = (int)(2u << 16);                                      // data_size=4B, mask=0
    g1[1] = (int)((tensorW & 0xFFFFu) << 16);                     // tensor_dim0 lo16
    g1[2] = (int)((tensorW >> 16) | ((tensorH & 0xFFFFu) << 16)); // dim0 hi | dim1 lo
    g1[3] = (int)((tensorH >> 16) | (tileW << 16));               // dim1 hi | tile_dim0
    g1[4] = (int)(tileH & 0xFFFFu);                               // tile_dim1 (tile_dim2=0)
    g1[5] = (int)rowStrideElems;                                  // tensor_dim0_stride lo32
    g1[6] = 0; g1[7] = 0;
    i32x4 gz = {0, 0, 0, 0};
#if __clang_major__ >= 23
    i32x8 gz8 = {0, 0, 0, 0, 0, 0, 0, 0};
    __builtin_amdgcn_tensor_load_to_lds(g0, g1, gz, gz, gz8, 0);
#else
    __builtin_amdgcn_tensor_load_to_lds(g0, g1, gz, gz, 0);
#endif
    __builtin_amdgcn_s_wait_tensorcnt(0);
#endif
}

// ---- LayerNorm over channels: x(B,C,H,W) -> xn(B*L, 192) -------------------
__global__ void ln_kernel(const float* __restrict__ x,
                          const float* __restrict__ g,
                          const float* __restrict__ bta,
                          float* __restrict__ xn)
{
    int pos  = blockIdx.x * (blockDim.x >> 5) + (threadIdx.x >> 5);
    int lane = threadIdx.x & 31;
    if (pos >= BLTOT) return;
    int b = pos >> 12, l = pos & (LSEQ - 1);
    const float* xp = x + (size_t)b * CDIM * LSEQ + l;
    float vals[6], s = 0.f, s2 = 0.f;
    #pragma unroll
    for (int i = 0; i < 6; ++i) {
        int c = lane + 32 * i;
        float v = xp[(size_t)c * LSEQ];
        vals[i] = v; s += v; s2 += v * v;
    }
    #pragma unroll
    for (int m = 16; m >= 1; m >>= 1) {
        s  += __shfl_xor(s,  m, 32);
        s2 += __shfl_xor(s2, m, 32);
    }
    float mu  = s * (1.f / CDIM);
    float var = s2 * (1.f / CDIM) - mu * mu;
    float r   = rsqrtf(var + 1e-5f);
    float* o = xn + (size_t)pos * CDIM;
    #pragma unroll
    for (int i = 0; i < 6; ++i) {
        int c = lane + 32 * i;
        o[c] = (vals[i] - mu) * r * g[c] + bta[c];
    }
}

// ---- Generic WMMA GEMM: C[M, N] = A[M, K] (* Ascale) @ B[K, N] -------------
// A: f32 in global, loaded b128 + cvt to bf16. B: staged to LDS by TDM in
// KSTAGE-row chunks, read via ds loads. f32 accumulation via
// v_wmma_f32_16x16x32_bf16. Each wave computes a 16 x (16*NT) strip.
// STOREMODE 0: C[m*ldc + n];  STOREMODE 1: C[((m>>12)*N + n)*LSEQ + (m&4095)]
template<int NT, int KSTAGE, bool SCALED, bool BOUNDED, int STOREMODE>
__global__ __launch_bounds__(256) void wmma_gemm(
        const float* __restrict__ A,
        const float* __restrict__ Ascale,
        const float* __restrict__ Bm,
        float* __restrict__ C,
        int M, int N, int K, int lda, int ldb, int ldc)
{
    constexpr int TW = 16 * NT;
    __shared__ float ldsB[KSTAGE * TW];

    const int lane   = threadIdx.x & 31;
    const int wave   = (blockIdx.x * blockDim.x + threadIdx.x) >> 5;
    const int mTiles = M >> 4;
    const int mTile  = wave % mTiles;   // grid sized exactly: no residue waves
    const int nGroup = wave / mTiles;   // uniform across the block (mTiles % 8 == 0)

    const int half  = lane >> 4;
    const int idx16 = lane & 15;
    const int mRow  = mTile * 16 + idx16;
    const int kbA   = half ? 8  : 0;    // ISA bf16 A 16x32 lane layout
    const int kbB   = half ? 8  : 0;    // ISA bf16 B 32x16 lane layout
    const int nColG = nGroup * TW;      // global column base of this strip

    const float* Arow = A + (size_t)mRow * lda;
    const float* Srow = SCALED ? (Ascale + (size_t)mRow * lda) : (const float*)0;

    v8f acc[NT];
    #pragma unroll
    for (int t = 0; t < NT; ++t) acc[t] = {};

    for (int ks = 0; ks < K; ks += KSTAGE) {
        __syncthreads();                       // previous stage fully consumed
#if HAVE_TDM
        if ((threadIdx.x >> 5) == 0) {
            tdm_load_2d_f32(Bm + (size_t)ks * ldb + nColG,
                            (unsigned)(unsigned long long)(uintptr_t)ldsB,
                            (unsigned)(N - nColG), (unsigned)(K - ks),
                            (unsigned)TW, (unsigned)KSTAGE, (unsigned)ldb);
        }
#else
        for (int idx = threadIdx.x; idx < KSTAGE * TW; idx += 256) {
            int kk = idx / TW, nn = idx % TW;
            int gn = nColG + nn;
            ldsB[idx] = (!BOUNDED || gn < N) ? Bm[(size_t)(ks + kk) * ldb + gn] : 0.f;
        }
#endif
        __syncthreads();

        for (int k0 = 0; k0 < KSTAGE; k0 += 32) {
            // A fragment: two contiguous 8-float runs -> 4x b128 loads
            const float* ap = Arow + ks + k0 + kbA;
            float4 a0 = *(const float4*)(ap);
            float4 a1 = *(const float4*)(ap + 4);
            float4 a2 = *(const float4*)(ap + 16);
            float4 a3 = *(const float4*)(ap + 20);
            if (SCALED) {
                const float* sp = Srow + ks + k0 + kbA;
                float4 s0 = *(const float4*)(sp);
                float4 s1 = *(const float4*)(sp + 4);
                float4 s2 = *(const float4*)(sp + 16);
                float4 s3 = *(const float4*)(sp + 20);
                a0.x *= s0.x; a0.y *= s0.y; a0.z *= s0.z; a0.w *= s0.w;
                a1.x *= s1.x; a1.y *= s1.y; a1.z *= s1.z; a1.w *= s1.w;
                a2.x *= s2.x; a2.y *= s2.y; a2.z *= s2.z; a2.w *= s2.w;
                a3.x *= s3.x; a3.y *= s3.y; a3.z *= s3.z; a3.w *= s3.w;
            }
            v16bf av;
            av[0]  = (__bf16)a0.x; av[1]  = (__bf16)a0.y; av[2]  = (__bf16)a0.z; av[3]  = (__bf16)a0.w;
            av[4]  = (__bf16)a1.x; av[5]  = (__bf16)a1.y; av[6]  = (__bf16)a1.z; av[7]  = (__bf16)a1.w;
            av[8]  = (__bf16)a2.x; av[9]  = (__bf16)a2.y; av[10] = (__bf16)a2.z; av[11] = (__bf16)a2.w;
            av[12] = (__bf16)a3.x; av[13] = (__bf16)a3.y; av[14] = (__bf16)a3.z; av[15] = (__bf16)a3.w;

            #pragma unroll
            for (int t = 0; t < NT; ++t) {
                const float* bp = ldsB + (size_t)(k0 + kbB) * TW + idx16 + 16 * t;
                v16bf bv;
                #pragma unroll
                for (int e = 0; e < 16; ++e)
                    bv[e] = (__bf16)bp[(e < 8 ? e : e + 8) * TW];     // ds loads, immediate offsets
                acc[t] = __builtin_amdgcn_wmma_f32_16x16x32_bf16(
                             false, av, false, bv, (short)0, acc[t], false, false);
            }
        }
    }

    const int mBase = mTile * 16 + (half ? 8 : 0);
    #pragma unroll
    for (int t = 0; t < NT; ++t) {
        const int nCol = nColG + idx16 + 16 * t;
        if (STOREMODE == 0) {
            if (!BOUNDED || nCol < N) {
                #pragma unroll
                for (int r = 0; r < 8; ++r)
                    C[(size_t)(mBase + r) * ldc + nCol] = acc[t][r];
            }
        } else {
            #pragma unroll
            for (int r = 0; r < 8; ++r) {
                int m = mBase + r;
                int b = m >> 12, l = m & (LSEQ - 1);
                C[((size_t)b * N + nCol) * LSEQ + l] = acc[t][r];
            }
        }
    }
}

// ---- causal (dir=0) / anti-causal (dir=1) depthwise conv + SiLU ------------
__global__ void conv_silu_kernel(const float* __restrict__ xz,
                                 const float* __restrict__ cw,
                                 const float* __restrict__ cb,
                                 float* __restrict__ xc, int dir)
{
    size_t i = (size_t)blockIdx.x * blockDim.x + threadIdx.x;
    if (i >= (size_t)BLTOT * DINNER) return;
    int d = (int)(i % DINNER);
    size_t pos = i / DINNER;
    int b = (int)(pos >> 12), l = (int)(pos & (LSEQ - 1));
    float acc = cb[d];
    #pragma unroll
    for (int k = 0; k < 4; ++k) {
        int t = dir ? (l + 3 - k) : (l - 3 + k);
        if (t >= 0 && t < LSEQ)
            acc += xz[((size_t)(b * LSEQ + t)) * (2 * DINNER) + d] * cw[d * 4 + k];
    }
    xc[pos * DINNER + d] = acc * (1.f / (1.f + __expf(-acc)));   // SiLU
}

// ---- silu(z): z half of xz -> sz(B*L, 384) ---------------------------------
__global__ void silu_z_kernel(const float* __restrict__ xz, float* __restrict__ sz)
{
    size_t i = (size_t)blockIdx.x * blockDim.x + threadIdx.x;
    if (i >= (size_t)BLTOT * DINNER) return;
    int d = (int)(i % DINNER);
    size_t pos = i / DINNER;
    float zv = xz[pos * (2 * DINNER) + DINNER + d];
    sz[pos * DINNER + d] = zv * (1.f / (1.f + __expf(-zv)));
}

// ---- dt = softplus(proj[:, :12] @ W_dt + b_dt) -----------------------------
__global__ void dt_kernel(const float* __restrict__ proj,
                          const float* __restrict__ Wdt,
                          const float* __restrict__ bdt,
                          float* __restrict__ dt)
{
    size_t i = (size_t)blockIdx.x * blockDim.x + threadIdx.x;
    if (i >= (size_t)BLTOT * DINNER) return;
    int d = (int)(i % DINNER);
    size_t pos = i / DINNER;
    const float* pr = proj + pos * PROJN;
    float acc = bdt[d];
    #pragma unroll
    for (int r = 0; r < DTRANK; ++r)
        acc = fmaf(pr[r], Wdt[r * DINNER + d], acc);
    dt[pos * DINNER + d] = (acc > 20.f) ? acc : log1pf(__expf(acc));
}

// ---- scan phase 1: per-chunk local end-state E and decay product P ---------
__global__ void scan_phase1(const float* __restrict__ xc,
                            const float* __restrict__ dt,
                            const float* __restrict__ proj,
                            const float* __restrict__ A_log,
                            float* __restrict__ Ebuf, float* __restrict__ Pbuf,
                            int dir)
{
    int b = blockIdx.x / NCHUNK, chunk = blockIdx.x % NCHUNK;
    int pair = blockIdx.y * blockDim.x + threadIdx.x;      // 0..6143
    int d = pair >> 4, s = pair & 15;
    float Ac = -__expf(A_log[d * DSTATE + s]);
    float h = 0.f, P = 1.f;
    for (int i = 0; i < CHUNK; ++i) {
        int l = dir ? (LSEQ - 1 - (chunk * CHUNK + i)) : (chunk * CHUNK + i);
        size_t pos = (size_t)b * LSEQ + l;
        float dtv = dt[pos * DINNER + d];
        float dA  = __expf(dtv * Ac);
        float Bv  = proj[pos * PROJN + DTRANK + s];
        float xv  = xc[pos * DINNER + d];
        h = fmaf(dA, h, dtv * Bv * xv);
        P *= dA;
    }
    size_t o = (((size_t)b * NCHUNK + chunk) * DINNER + d) * DSTATE + s;
    Ebuf[o] = h; Pbuf[o] = P;
}

// ---- scan phase 2: compose chunk summaries -> chunk-start states -----------
__global__ void scan_phase2(const float* __restrict__ Ebuf,
                            const float* __restrict__ Pbuf,
                            float* __restrict__ Hinit)
{
    int idx = blockIdx.x * blockDim.x + threadIdx.x;
    if (idx >= NBATCH * DINNER * DSTATE) return;
    int b = idx / (DINNER * DSTATE), p = idx % (DINNER * DSTATE);
    float h = 0.f;
    for (int c = 0; c < NCHUNK; ++c) {
        size_t o = ((size_t)b * NCHUNK + c) * (DINNER * DSTATE) + p;
        Hinit[o] = h;
        h = fmaf(Pbuf[o], h, Ebuf[o]);
    }
}

// ---- scan phase 3: replay with correct h0, reduce over states, emit y ------
__global__ void scan_phase3(const float* __restrict__ xc,
                            const float* __restrict__ dt,
                            const float* __restrict__ proj,
                            const float* __restrict__ A_log,
                            const float* __restrict__ Dp,
                            const float* __restrict__ Hinit,
                            float* __restrict__ y, int dir)
{
    int b = blockIdx.x / NCHUNK, chunk = blockIdx.x % NCHUNK;
    int s = threadIdx.x & 15;
    int d = blockIdx.y * 16 + (threadIdx.x >> 4);
    float Ac = -__expf(A_log[d * DSTATE + s]);
    float Dv = Dp[d];
    size_t hio = (((size_t)b * NCHUNK + chunk) * DINNER + d) * DSTATE + s;
    float h = Hinit[hio];
    for (int i = 0; i < CHUNK; ++i) {
        int l = dir ? (LSEQ - 1 - (chunk * CHUNK + i)) : (chunk * CHUNK + i);
        size_t pos = (size_t)b * LSEQ + l;
        float dtv = dt[pos * DINNER + d];
        float dA  = __expf(dtv * Ac);
        float Bv  = proj[pos * PROJN + DTRANK + s];
        float Cv  = proj[pos * PROJN + DTRANK + DSTATE + s];
        float xv  = xc[pos * DINNER + d];
        h = fmaf(dA, h, dtv * Bv * xv);
        float yv = h * Cv;
        yv += __shfl_xor(yv, 1, 32);
        yv += __shfl_xor(yv, 2, 32);
        yv += __shfl_xor(yv, 4, 32);
        yv += __shfl_xor(yv, 8, 32);
        if (s == 0) {
            float out = fmaf(Dv, xv, yv);
            size_t yo = pos * DINNER + d;
            if (dir == 0) y[yo] = out; else y[yo] += out;
        }
    }
}

// ---- launch ----------------------------------------------------------------
extern "C" void kernel_launch(void* const* d_in, const int* in_sizes, int n_in,
                              void* d_out, int out_size, void* d_ws, size_t ws_size,
                              hipStream_t stream)
{
    const float* x      = (const float*)d_in[0];
    const float* ln_g   = (const float*)d_in[1];
    const float* ln_b   = (const float*)d_in[2];
    const float* W_in   = (const float*)d_in[3];
    const float* conv_w = (const float*)d_in[4];
    const float* conv_b = (const float*)d_in[5];
    const float* W_xprj = (const float*)d_in[6];
    const float* W_dt   = (const float*)d_in[7];
    const float* b_dt   = (const float*)d_in[8];
    const float* A_log  = (const float*)d_in[9];
    const float* Dp     = (const float*)d_in[10];
    const float* W_out  = (const float*)d_in[11];
    float* out = (float*)d_out;

    // workspace carve-up (floats)
    float* w = (float*)d_ws;
    float* xn   = w;                                  // 16384*192
    float* xz   = xn   + (size_t)BLTOT * CDIM;        // 16384*768
    float* xc   = xz   + (size_t)BLTOT * 2 * DINNER;  // 16384*384
    float* sz   = xc   + (size_t)BLTOT * DINNER;      // 16384*384
    float* proj = sz   + (size_t)BLTOT * DINNER;      // 16384*44
    float* dt   = proj + (size_t)BLTOT * PROJN;       // 16384*384
    float* yb   = dt   + (size_t)BLTOT * DINNER;      // 16384*384
    float* Eb   = yb   + (size_t)BLTOT * DINNER;      // 4*16*384*16
    float* Pb   = Eb   + (size_t)NBATCH * NCHUNK * DINNER * DSTATE;
    float* Hi   = Pb   + (size_t)NBATCH * NCHUNK * DINNER * DSTATE;

    const int TPB = 256;
    const size_t NEL = (size_t)BLTOT * DINNER;        // 6,291,456
    const int ELBLK = (int)((NEL + TPB - 1) / TPB);

    // 1) LayerNorm
    ln_kernel<<<BLTOT / 8, TPB, 0, stream>>>(x, ln_g, ln_b, xn);

    // 2) xz = xn @ W_in   (M=16384, N=768, K=192): 1024 * 12 wave-strips
    {
        int waves = (BLTOT / 16) * (768 / 64);
        wmma_gemm<4, 192, false, false, 0><<<waves / 8, TPB, 0, stream>>>(
            xn, nullptr, W_in, xz, BLTOT, 2 * DINNER, CDIM,
            CDIM, 2 * DINNER, 2 * DINNER);
    }

    // 3) sz = silu(z)
    silu_z_kernel<<<ELBLK, TPB, 0, stream>>>(xz, sz);

    for (int dir = 0; dir < 2; ++dir) {
        // 4a) xc = silu(conv_dir(x))
        conv_silu_kernel<<<ELBLK, TPB, 0, stream>>>(xz, conv_w, conv_b, xc, dir);

        // 4b) proj = xc @ W_xproj  (M=16384, N=44 -> one 3-tile strip, K=384)
        {
            int waves = (BLTOT / 16) * 1;
            wmma_gemm<3, 192, false, true, 0><<<waves / 8, TPB, 0, stream>>>(
                xc, nullptr, W_xprj, proj, BLTOT, PROJN, DINNER,
                DINNER, PROJN, PROJN);
        }

        // 4c) dt = softplus(proj[:, :12] @ W_dt + b_dt)
        dt_kernel<<<ELBLK, TPB, 0, stream>>>(proj, W_dt, b_dt, dt);

        // 4d-f) chunked selective scan
        dim3 g1(NBATCH * NCHUNK, (DINNER * DSTATE) / TPB);   // (64, 24)
        scan_phase1<<<g1, TPB, 0, stream>>>(xc, dt, proj, A_log, Eb, Pb, dir);
        scan_phase2<<<(NBATCH * DINNER * DSTATE + TPB - 1) / TPB, TPB, 0, stream>>>(Eb, Pb, Hi);
        dim3 g3(NBATCH * NCHUNK, DINNER / 16);               // (64, 24)
        scan_phase3<<<g3, TPB, 0, stream>>>(xc, dt, proj, A_log, Dp, Hi, yb, dir);
    }

    // 5) out(B,C,H,W) = (y_comb * sz) @ W_out, transposed store
    {
        int waves = (BLTOT / 16) * (CDIM / 64);
        wmma_gemm<4, 192, true, false, 1><<<waves / 8, TPB, 0, stream>>>(
            yb, sz, W_out, out, BLTOT, CDIM, DINNER,
            DINNER, CDIM, CDIM);
    }
}
